// FWBlock_24223615549825
// MI455X (gfx1250) — matrix operand order repaired and checked by reading, not compiled, not run on hardware
//
#include <hip/hip_runtime.h>
#include <hip/hip_bf16.h>

// ---------------------------------------------------------------------------
// Problem constants (from setup_inputs): B=2, L=2048, D=1024, V=8192, F=4D,
// n_chunks=16 -> C=128.  All fp32 in/out.
// ---------------------------------------------------------------------------
#define BB   2
#define LL   2048
#define DD   1024
#define FF   4096
#define VV   8192
#define NC   16
#define CC   128          // L / n_chunks
#define BL   (BB*LL)      // 4096 rows

typedef __attribute__((ext_vector_type(16))) __bf16 v16bf;
typedef __attribute__((ext_vector_type(8)))  float  v8f;

// pack two floats as bf16x2 in one dword: hardware RNE pack (1 VALU op)
__device__ __forceinline__ unsigned pack2(float a, float b) {
  unsigned r;
  asm("v_cvt_pk_bf16_f32 %0, %1, %2" : "=v"(r) : "v"(a), "v"(b));
  return r;
}

// LDS row padding: 36 halves = 72B -> 8B aligned at k%4==0, conflict-free frags
#define LPAD 36

// ---------------------------------------------------------------------------
// Stage one 64x32 A tile and one 32x128 B tile (fp32 -> bf16) into LDS.
//   TA: A stored (K,M) row-major.  TB: B stored (N,K) row-major.
// Two-phase: issue ALL global_load_b128s first (max memory-level parallelism),
// then convert (v_cvt_pk_bf16_f32) + packed LDS stores.
// ---------------------------------------------------------------------------
template<bool TA, bool TB>
__device__ __forceinline__ void stage_tiles(
    const float* __restrict__ A, const float* __restrict__ B,
    int k0, int m0, int n0, int lda, int ldb,
    unsigned short (*sA)[LPAD], unsigned short (*sB)[LPAD], int t)
{
  float4 av[2], bv[4];

  // ---------------- phase 1: issue all global loads ----------------
  if (TA) {
    // contiguous along M: two adjacent k-rows, 4 m's per float4
    const int kp = t >> 4;            // 0..15 -> k = 2*kp
    const int mq = (t & 15) * 4;
    const int k  = 2 * kp;
    const float* p0 = &A[(long)(k0 + k) * lda + m0 + mq];
    av[0] = *(const float4*)p0;
    av[1] = *(const float4*)(p0 + lda);
  } else {
    #pragma unroll
    for (int j = 0; j < 2; ++j) {
      const int i = t + j * 256;
      const int row = i >> 3, kq = (i & 7) * 4;
      av[j] = *(const float4*)&A[(long)(m0 + row) * lda + k0 + kq];
    }
  }
  if (TB) {
    #pragma unroll
    for (int j = 0; j < 4; ++j) {
      const int i = t + j * 256;
      const int n = i >> 3, kq = (i & 7) * 4;
      bv[j] = *(const float4*)&B[(long)(n0 + n) * ldb + k0 + kq];
    }
  } else {
    #pragma unroll
    for (int j = 0; j < 2; ++j) {
      const int i = t + j * 256;
      const int kp = i >> 5;          // 0..15 -> k = 2*kp
      const int nq = (i & 31) * 4;
      const int k  = 2 * kp;
      const float* p0 = &B[(long)(k0 + k) * ldb + n0 + nq];
      bv[2 * j]     = *(const float4*)p0;
      bv[2 * j + 1] = *(const float4*)(p0 + ldb);
    }
  }

  // ---------------- phase 2: convert + LDS store ----------------
  if (TA) {
    const int kp = t >> 4;
    const int mq = (t & 15) * 4;
    const int k  = 2 * kp;
    *(unsigned*)&sA[mq + 0][k] = pack2(av[0].x, av[1].x);
    *(unsigned*)&sA[mq + 1][k] = pack2(av[0].y, av[1].y);
    *(unsigned*)&sA[mq + 2][k] = pack2(av[0].z, av[1].z);
    *(unsigned*)&sA[mq + 3][k] = pack2(av[0].w, av[1].w);
  } else {
    #pragma unroll
    for (int j = 0; j < 2; ++j) {
      const int i = t + j * 256;
      const int row = i >> 3, kq = (i & 7) * 4;
      uint2 p; p.x = pack2(av[j].x, av[j].y); p.y = pack2(av[j].z, av[j].w);
      *(uint2*)&sA[row][kq] = p;
    }
  }
  if (TB) {
    #pragma unroll
    for (int j = 0; j < 4; ++j) {
      const int i = t + j * 256;
      const int n = i >> 3, kq = (i & 7) * 4;
      uint2 p; p.x = pack2(bv[j].x, bv[j].y); p.y = pack2(bv[j].z, bv[j].w);
      *(uint2*)&sB[n][kq] = p;
    }
  } else {
    #pragma unroll
    for (int j = 0; j < 2; ++j) {
      const int i = t + j * 256;
      const int kp = i >> 5;
      const int nq = (i & 31) * 4;
      const int k  = 2 * kp;
      const float4 b0 = bv[2 * j], b1 = bv[2 * j + 1];
      *(unsigned*)&sB[nq + 0][k] = pack2(b0.x, b1.x);
      *(unsigned*)&sB[nq + 1][k] = pack2(b0.y, b1.y);
      *(unsigned*)&sB[nq + 2][k] = pack2(b0.z, b1.z);
      *(unsigned*)&sB[nq + 3][k] = pack2(b0.w, b1.w);
    }
  }
}

// ---------------------------------------------------------------------------
// Tiled bf16-WMMA GEMM with fp32 accumulate, LDS double-buffered.
//   D = A*B (+bias) (+D if accumulate) ; optional strict-lower mask (keep n<m)
// Tile 64x128x32, 256 threads = 8 waves; each wave owns a 32x32 output slab
// (2 A-frags x 2 B-frags -> 4 v_wmma_f32_16x16x32_bf16 per K-step).
// Fragments use contiguous k = 16*(lane/16) + s for both A and B (wave32).
// ---------------------------------------------------------------------------
template<bool TA, bool TB>
__global__ __launch_bounds__(256) void gemm_wmma_bf16(
    const float* __restrict__ Ab, const float* __restrict__ Bb,
    float* __restrict__ Db, const float* __restrict__ bias,
    int M, int N, int K, int lda, int ldb, int ldd,
    long strideA, long strideB, long strideD,
    int accumulate, int maskStrict)
{
  const float* A = Ab + (long)blockIdx.z * strideA;
  const float* B = Bb + (long)blockIdx.z * strideB;
  float*       Dm = Db + (long)blockIdx.z * strideD;

  __shared__ unsigned short lA[2][64][LPAD];   // [buf][m][k]
  __shared__ unsigned short lB[2][128][LPAD];  // [buf][n][k]

  const int t    = threadIdx.x;
  const int wave = t >> 5;
  const int lane = t & 31;
  const int wm   = (wave & 1) * 32;   // wave row offset inside tile
  const int wn   = (wave >> 1) * 32;  // wave col offset inside tile
  const int m0   = blockIdx.y * 64;
  const int n0   = blockIdx.x * 128;

  v8f acc[2][2] = {};

  const int lrow  = lane & 15;
  const int lkb8  = (lane & 16) ? 8 : 0;    // unused; A frag uses lkb16 too
  const int lkb16 = (lane & 16) ? 16 : 0;   // B frag K-phase

  const int nk = K >> 5;
  stage_tiles<TA, TB>(A, B, 0, m0, n0, lda, ldb, lA[0], lB[0], t);
  __syncthreads();

  for (int ik = 0; ik < nk; ++ik) {
    const int cur = ik & 1;
    if (ik + 1 < nk)
      stage_tiles<TA, TB>(A, B, (ik + 1) * 32, m0, n0, lda, ldb,
                          lA[cur ^ 1], lB[cur ^ 1], t);
    if (ik + 2 < nk) {  // speculative prefetch two tiles ahead (A operand)
      const float* pa = TA ? &A[(long)((ik + 2) * 32 + (t >> 6)) * lda + m0 + (t & 63)]
                           : &A[(long)(m0 + (t >> 2)) * lda + (ik + 2) * 32];
      __builtin_prefetch(pa, 0, 1);
    }

    // ---- B fragments: lane holds col n, K = lkb16..lkb16+15 ----
    union { unsigned u[8]; v16bf v; } bfr[2];
    #pragma unroll
    for (int ni = 0; ni < 2; ++ni) {
      const unsigned* s = (const unsigned*)&lB[cur][wn + ni * 16 + (lane & 15)][lkb16];
      #pragma unroll
      for (int j = 0; j < 8; ++j) bfr[ni].u[j] = s[j];
    }
    // ---- A fragments + 4 WMMAs ----
    #pragma unroll
    for (int mi = 0; mi < 2; ++mi) {
      union { unsigned u[8]; v16bf v; } afr;
      const int row = wm + mi * 16 + lrow;
      #pragma unroll
      for (int vg = 0; vg < 8; ++vg) {
        const int kk = lkb16 + 2 * vg;
        afr.u[vg] = *(const unsigned*)&lA[cur][row][kk];
      }
      #pragma unroll
      for (int ni = 0; ni < 2; ++ni)
        acc[mi][ni] = __builtin_amdgcn_wmma_f32_16x16x32_bf16(
            false, afr.v, false, bfr[ni].v, (short)0, acc[mi][ni], false, false);
    }
    __syncthreads();
  }

  // ---- epilogue: C/D layout -> lane col = lane&15, rows rbase..rbase+7 ----
  const int rbase = (lane & 16) ? 8 : 0;
  #pragma unroll
  for (int ni = 0; ni < 2; ++ni) {
    const int nG = n0 + wn + ni * 16 + (lane & 15);
    const float bv = bias ? bias[nG] : 0.0f;
    #pragma unroll
    for (int mi = 0; mi < 2; ++mi) {
      #pragma unroll
      for (int i = 0; i < 8; ++i) {
        const int mG = m0 + wm + mi * 16 + rbase + i;
        float v = acc[mi][ni][i] + bv;
        if (accumulate) v += Dm[(long)mG * ldd + nG];
        if (maskStrict && (nG >= mG)) v = 0.0f;   // keep strictly-lower (n < m)
        Dm[(long)mG * ldd + nG] = v;
      }
    }
  }
}

// ---------------------------------------------------------------------------
// Block reductions (wave32)
// ---------------------------------------------------------------------------
__device__ __forceinline__ float blk_sum(float v, float* sbuf) {
  #pragma unroll
  for (int o = 16; o; o >>= 1) v += __shfl_down(v, o, 32);
  const int w = threadIdx.x >> 5;
  if ((threadIdx.x & 31) == 0) sbuf[w] = v;
  __syncthreads();
  float r = (threadIdx.x < (blockDim.x >> 5)) ? sbuf[threadIdx.x] : 0.0f;
  if (w == 0) {
    #pragma unroll
    for (int o = 16; o; o >>= 1) r += __shfl_down(r, o, 32);
    if (threadIdx.x == 0) sbuf[0] = r;
  }
  __syncthreads();
  r = sbuf[0];
  __syncthreads();
  return r;
}

__device__ __forceinline__ float blk_max(float v, float* sbuf) {
  #pragma unroll
  for (int o = 16; o; o >>= 1) v = fmaxf(v, __shfl_down(v, o, 32));
  const int w = threadIdx.x >> 5;
  if ((threadIdx.x & 31) == 0) sbuf[w] = v;
  __syncthreads();
  float r = (threadIdx.x < (blockDim.x >> 5)) ? sbuf[threadIdx.x] : -3.4e38f;
  if (w == 0) {
    #pragma unroll
    for (int o = 16; o; o >>= 1) r = fmaxf(r, __shfl_down(r, o, 32));
    if (threadIdx.x == 0) sbuf[0] = r;
  }
  __syncthreads();
  r = sbuf[0];
  __syncthreads();
  return r;
}

// ---------------------------------------------------------------------------
// Elementwise / rowwise kernels
// ---------------------------------------------------------------------------
__global__ void k_sqrelu(const float* __restrict__ in, float* __restrict__ out, long n) {
  long i = (long)blockIdx.x * blockDim.x + threadIdx.x;
  const long st = (long)gridDim.x * blockDim.x;
  for (; i < n; i += st) { float r = fmaxf(in[i], 0.0f); out[i] = r * r; }
}

// x2 = sqrelu(h1 - steps[sidx]*att), in place on h1
__global__ void k_combine_sqrelu(float* __restrict__ h, const float* __restrict__ att,
                                 const float* __restrict__ steps, int sidx, long n) {
  const float s = steps[sidx];
  long i = (long)blockIdx.x * blockDim.x + threadIdx.x;
  const long st = (long)gridDim.x * blockDim.x;
  for (; i < n; i += st) { float a = h[i] - s * att[i]; float r = fmaxf(a, 0.0f); h[i] = r * r; }
}

__global__ void k_axpy_neg(float* __restrict__ x, const float* __restrict__ y,
                           const float* __restrict__ steps, int sidx, long n) {
  const float s = steps[sidx];
  long i = (long)blockIdx.x * blockDim.x + threadIdx.x;
  const long st = (long)gridDim.x * blockDim.x;
  for (; i < n; i += st) x[i] -= s * y[i];
}

__global__ void k_mul(const float* __restrict__ a, const float* __restrict__ b,
                      float* __restrict__ out, long n) {
  long i = (long)blockIdx.x * blockDim.x + threadIdx.x;
  const long st = (long)gridDim.x * blockDim.x;
  for (; i < n; i += st) out[i] = a[i] * b[i];
}

// g1 = g2 * 2*relu(h1), in place on g
__global__ void k_gradrelu(float* __restrict__ g, const float* __restrict__ h1, long n) {
  long i = (long)blockIdx.x * blockDim.x + threadIdx.x;
  const long st = (long)gridDim.x * blockDim.x;
  for (; i < n; i += st) g[i] *= 2.0f * fmaxf(h1[i], 0.0f);
}

__global__ void k_reduce_sum(const float* __restrict__ w, float* __restrict__ out, int n) {
  __shared__ float sbuf[8];
  float s = 0.0f;
  for (int i = threadIdx.x; i < n; i += blockDim.x) s += w[i];
  s = blk_sum(s, sbuf);
  if (threadIdx.x == 0) out[0] = s;
}

// LayerNorm forward.  cumS==null: slow mode (vec scale/bias, writes norm & rstd)
// cumS!=null: fast mode: scale = svec - steps[sidx]*cumS[row], bias likewise.
__global__ __launch_bounds__(256) void k_ln_fwd(
    const float* __restrict__ X, float* __restrict__ Y,
    float* __restrict__ normOut, float* __restrict__ rstdOut,
    const float* __restrict__ svec, const float* __restrict__ bvec,
    const float* __restrict__ cumS, const float* __restrict__ cumB,
    const float* __restrict__ steps, int sidx, int Dn)
{
  __shared__ float sbuf[8];
  const long row = blockIdx.x;
  const float* x = X + row * Dn;
  float s = 0.0f, ss = 0.0f;
  for (int d = threadIdx.x; d < Dn; d += blockDim.x) { float v = x[d]; s += v; ss += v * v; }
  s  = blk_sum(s,  sbuf);
  ss = blk_sum(ss, sbuf);
  const float mu   = s / Dn;
  const float var  = ss / Dn - mu * mu;
  const float rstd = rsqrtf(var + 1e-6f);
  if (rstdOut && threadIdx.x == 0) rstdOut[row] = rstd;
  const float st = cumS ? steps[sidx] : 0.0f;
  for (int d = threadIdx.x; d < Dn; d += blockDim.x) {
    const float nv = (x[d] - mu) * rstd;
    if (normOut) normOut[row * Dn + d] = nv;
    float sc = svec[d], bi = bvec[d];
    if (cumS) { sc -= st * cumS[row * Dn + d]; bi -= st * cumB[row * Dn + d]; }
    Y[row * Dn + d] = nv * sc + bi;
  }
}

// LayerNorm backward: g3 = rstd*(dy - mean(dy) - norm*mean(dy*norm)), dy=g4*scale
__global__ __launch_bounds__(256) void k_ln_bwd(
    const float* __restrict__ g4, const float* __restrict__ norm,
    const float* __restrict__ rstd, const float* __restrict__ svec,
    float* __restrict__ g3, int Dn)
{
  __shared__ float sbuf[8];
  const long row = blockIdx.x;
  const float* g  = g4   + row * Dn;
  const float* nm = norm + row * Dn;
  float s1 = 0.0f, s2 = 0.0f;
  for (int d = threadIdx.x; d < Dn; d += blockDim.x) {
    const float dy = g[d] * svec[d]; s1 += dy; s2 += dy * nm[d];
  }
  s1 = blk_sum(s1, sbuf);
  s2 = blk_sum(s2, sbuf);
  const float m1 = s1 / Dn, m2 = s2 / Dn, r = rstd[row];
  for (int d = threadIdx.x; d < Dn; d += blockDim.x) {
    const float dy = g[d] * svec[d];
    g3[row * Dn + d] = r * (dy - m1 - nm[d] * m2);
  }
}

// softmax-CE grad, in place on logits:
// g = (w_row/max(wsum,1e-8)) * (softmax * sum(labels) - labels)
__global__ __launch_bounds__(256) void k_ce_grad(
    float* __restrict__ logits, const float* __restrict__ labels,
    const float* __restrict__ weights, const float* __restrict__ wsum, int Vn)
{
  __shared__ float sbuf[8];
  const long row = blockIdx.x;
  float* lg = logits + row * (long)Vn;
  const float* lb = labels + row * (long)Vn;
  float mx = -3.4e38f;
  for (int j = threadIdx.x; j < Vn; j += blockDim.x) mx = fmaxf(mx, lg[j]);
  mx = blk_max(mx, sbuf);
  float se = 0.0f, ls = 0.0f;
  for (int j = threadIdx.x; j < Vn; j += blockDim.x) { se += expf(lg[j] - mx); ls += lb[j]; }
  se = blk_sum(se, sbuf);
  ls = blk_sum(ls, sbuf);
  const float scale = weights[row] / fmaxf(wsum[0], 1e-8f);
  const float inv = 1.0f / se;
  for (int j = threadIdx.x; j < Vn; j += blockDim.x) {
    const float p = expf(lg[j] - mx) * inv;
    lg[j] = scale * (p * ls - lb[j]);
  }
}

// exclusive cumsum over L axis of a (B,L,Dn) array; 1 thread per (b,d)
__global__ void k_excl_cumsum_L(const float* __restrict__ in, float* __restrict__ out,
                                int Bn, int Ln, int Dn)
{
  const int idx = blockIdx.x * blockDim.x + threadIdx.x;
  if (idx >= Bn * Dn) return;
  const int b = idx / Dn, d = idx % Dn;
  const long base = (long)b * Ln * Dn + d;
  float run = 0.0f;
  for (int l = 0; l < Ln; ++l) {
    const long p = base + (long)l * Dn;
    const float v = in[p];
    out[p] = run;
    run += v;
  }
}

// ---------------------------------------------------------------------------
// Host-side launch helpers
// ---------------------------------------------------------------------------
static inline void launch_gemm(hipStream_t st, bool ta, bool tb,
                               const float* A, const float* B, float* Dp, const float* bias,
                               int M, int N, int K, int lda, int ldb, int ldd,
                               long sA, long sB, long sD, int batch,
                               int accum, int mask)
{
  dim3 g(N / 128, M / 64, batch), b(256);
  if (!ta && !tb) gemm_wmma_bf16<false,false><<<g,b,0,st>>>(A,B,Dp,bias,M,N,K,lda,ldb,ldd,sA,sB,sD,accum,mask);
  else if (!ta &&  tb) gemm_wmma_bf16<false,true ><<<g,b,0,st>>>(A,B,Dp,bias,M,N,K,lda,ldb,ldd,sA,sB,sD,accum,mask);
  else if ( ta && !tb) gemm_wmma_bf16<true ,false><<<g,b,0,st>>>(A,B,Dp,bias,M,N,K,lda,ldb,ldd,sA,sB,sD,accum,mask);
  else                 gemm_wmma_bf16<true ,true ><<<g,b,0,st>>>(A,B,Dp,bias,M,N,K,lda,ldb,ldd,sA,sB,sD,accum,mask);
}

static inline int ew_blocks(long n) {
  long b = (n + 255) / 256;
  return (int)(b > 32768 ? 32768 : b);
}

// Mixed-chunk linear attention, sequential chunk-KV recurrence (KV is L2-resident):
//   for g: glob_g = q_g @ KV ; qk = tril(q_g k_g^T,-1) ; att_g = glob_g + qk v_g ;
//          KV += k_g^T v_g
static void run_attention(hipStream_t st, const float* q, const float* k, const float* v,
                          float* att, float* KV, float* QK, int S, int E)
{
  hipMemsetAsync(KV, 0, (size_t)BB * S * E * sizeof(float), st);
  const long qS = (long)LL * S, vS = (long)LL * E, kvS = (long)S * E;
  for (int g = 0; g < NC; ++g) {
    const float* qg = q + (long)g * CC * S;
    const float* kg = k + (long)g * CC * S;
    const float* vg = v + (long)g * CC * E;
    float* attg = att + (long)g * CC * E;
    // global part: att_g = q_g @ KV   (exclusive prefix state)
    launch_gemm(st, false, false, qg, KV, attg, nullptr,
                CC, E, S, S, E, E, qS, kvS, vS, BB, 0, 0);
    // local quadratic: QK = tril(q_g @ k_g^T, -1)
    launch_gemm(st, false, true, qg, kg, QK, nullptr,
                CC, CC, S, S, S, CC, qS, qS, (long)CC * CC, BB, 0, 1);
    // att_g += QK @ v_g
    launch_gemm(st, false, false, QK, vg, attg, nullptr,
                CC, E, CC, CC, E, E, (long)CC * CC, vS, vS, BB, 1, 0);
    // KV += k_g^T @ v_g
    launch_gemm(st, true, false, kg, vg, KV, nullptr,
                S, E, CC, S, E, E, qS, vS, kvS, BB, 1, 0);
  }
}

// ---------------------------------------------------------------------------
extern "C" void kernel_launch(void* const* d_in, const int* in_sizes, int n_in,
                              void* d_out, int out_size, void* d_ws, size_t ws_size,
                              hipStream_t stream)
{
  (void)in_sizes; (void)n_in; (void)out_size; (void)ws_size;
  const float* x        = (const float*)d_in[0];
  const float* labels   = (const float*)d_in[1];
  const float* weights  = (const float*)d_in[2];
  const float* W1       = (const float*)d_in[3];
  const float* b1       = (const float*)d_in[4];
  const float* W2       = (const float*)d_in[5];
  const float* b2       = (const float*)d_in[6];
  const float* ln_scale = (const float*)d_in[7];
  const float* ln_bias  = (const float*)d_in[8];
  const float* Wu       = (const float*)d_in[9];
  const float* bu       = (const float*)d_in[10];
  const float* steps    = (const float*)d_in[11];
  float* out = (float*)d_out;

  // workspace carve-up (floats)
  float* w = (float*)d_ws;
  size_t o = 0;
  const size_t BLF = (size_t)BL * FF, BLD = (size_t)BL * DD, BLV = (size_t)BL * VV;
  float* h1     = w + o; o += BLF;          // h1, later x1/x2
  float* h2     = w + o; o += BLF;          // slow h2 (attn2 keys)
  float* h3     = w + o; o += BLD;          // h3, later x3
  float* norm3  = w + o; o += BLD;
  float* h4     = w + o; o += BLD;
  float* rstd   = w + o; o += BL;
  float* logits = w + o; o += BLV;          // logits, then g_logits in place
  float* g4     = w + o; o += BLD;          // g4, later x4
  float* g3     = w + o; o += BLD;
  float* G      = w + o; o += BLF;          // g2, then g1 in place
  float* ATT    = w + o; o += BLF;          // attn1 out; attn2 reuses first BLD
  float* gscale = w + o; o += BLD;
  float* cumS   = w + o; o += BLD;
  float* cumB   = w + o; o += BLD;
  float* KV     = w + o; o += (size_t)BB * DD * FF;   // running chunk-KV state
  float* QK     = w + o; o += (size_t)BB * CC * CC;
  float* wsum   = w + o; o += 1;

  // ---------------- slow forward ----------------
  launch_gemm(stream,false,false, x,  W1, h1, b1, BL, FF, DD, DD, FF, FF, 0,0,0, 1, 0,0);
  k_sqrelu<<<ew_blocks(BLF),256,0,stream>>>(h1, h2, (long)BLF);
  launch_gemm(stream,false,false, h2, W2, h3, b2, BL, DD, FF, FF, DD, DD, 0,0,0, 1, 0,0);
  k_ln_fwd<<<BL,256,0,stream>>>(h3, h4, norm3, rstd, ln_scale, ln_bias,
                                nullptr, nullptr, steps, 0, DD);
  launch_gemm(stream,false,false, h4, Wu, logits, bu, BL, VV, DD, DD, VV, VV, 0,0,0, 1, 0,0);

  // ---------------- loss grads ----------------
  k_reduce_sum<<<1,256,0,stream>>>(weights, wsum, BL);
  k_ce_grad<<<BL,256,0,stream>>>(logits, labels, weights, wsum, VV);
  // g4 = g_logits @ Wu^T   (Wu stored (D,V) = (N,K) -> TB)
  launch_gemm(stream,false,true, logits, Wu, g4, nullptr, BL, DD, VV, VV, VV, DD, 0,0,0, 1, 0,0);
  k_mul<<<ew_blocks(BLD),256,0,stream>>>(g4, norm3, gscale, (long)BLD);
  k_ln_bwd<<<BL,256,0,stream>>>(g4, norm3, rstd, ln_scale, g3, DD);
  // g2 = g3 @ W2^T  (W2 stored (F,D) = (N,K) -> TB)
  launch_gemm(stream,false,true, g3, W2, G, nullptr, BL, FF, DD, DD, DD, FF, 0,0,0, 1, 0,0);
  k_gradrelu<<<ew_blocks(BLF),256,0,stream>>>(G, h1, (long)BLF);   // g1

  // ---------------- attention 1: q=x, k=x, v=g1 (S=D, E=F) ----------------
  run_attention(stream, x, x, G, ATT, KV, QK, DD, FF);
  // x2 = sqrelu(h1 - step0*attn1), in place on h1
  k_combine_sqrelu<<<ew_blocks(BLF),256,0,stream>>>(h1, ATT, steps, 0, (long)BLF);

  // ---------------- attention 2: q=x2, k=h2_slow, v=g3 (S=F, E=D) --------
  launch_gemm(stream,false,false, h1, W2, h3, b2, BL, DD, FF, FF, DD, DD, 0,0,0, 1, 0,0);
  run_attention(stream, h1, h2, g3, ATT, KV, QK, FF, DD);
  k_axpy_neg<<<ew_blocks(BLD),256,0,stream>>>(h3, ATT, steps, 2, (long)BLD);  // x3

  // ---------------- fast LN + unembed ----------------
  k_excl_cumsum_L<<<(BB*DD+255)/256,256,0,stream>>>(gscale, cumS, BB, LL, DD);
  k_excl_cumsum_L<<<(BB*DD+255)/256,256,0,stream>>>(g4,     cumB, BB, LL, DD);
  // x4 = norm(x3)*(ln_scale - step3*cumS) + (ln_bias - step3*cumB) -> reuse g4
  k_ln_fwd<<<BL,256,0,stream>>>(h3, g4, nullptr, nullptr, ln_scale, ln_bias,
                                cumS, cumB, steps, 3, DD);
  launch_gemm(stream,false,false, g4, Wu, out, bu, BL, VV, DD, DD, VV, VV, 0,0,0, 1, 0,0);
}
